// LearnedRecencyModel_35588099014933
// MI455X (gfx1250) — matrix operand order repaired and checked by reading, not compiled, not run on hardware
//
#include <hip/hip_runtime.h>
#include <hip/hip_bf16.h>

// ---------------------------------------------------------------------------
// LearnedRecencyModel on MI455X (gfx1250, wave32, WMMA)
//   V=32000 H=512 L=1024 B=64 HALF=256 PE=16
// Phase A: gather+MLP+LN+projections via v_wmma_f32_16x16x32_f16 (f16 in, f32 acc)
// Phase B: 1023-step delta-rule scan; 260KB padded state matrix per WG in LDS,
//          float4 (ds_*_b128) row traffic + software-pipelined k loads
// Phase C: 64x512x32000 output GEMM via WMMA on f16-transposed out_w
// ---------------------------------------------------------------------------

typedef __attribute__((ext_vector_type(16))) _Float16 v16h;
typedef __attribute__((ext_vector_type(8)))  float    v8f;

#define Vv 32000
#define Hh 512
#define Ll 1024
#define Bb 64
#define HALF 256
#define MSTRIDE 260   // padded row stride (floats): 1040 B, 16B aligned, 2-way-floor banks

// ---- WMMA helpers ----------------------------------------------------------

// A-matrix 16x32 f16 tile: lanes 0-15 row M=lane hold K k0..k0+15;
// lanes 16-31 same rows hold K k0+16..k0+31.
__device__ __forceinline__ v16h load_a_f16(const _Float16* base, int ld,
                                           int r0, int k0, int lane) {
  const int hf = lane >> 4, r = lane & 15;
  const _Float16* p = base + (size_t)(r0 + r) * ld + k0 + hf * 16;
  v16h v;
#pragma unroll
  for (int j = 0; j < 16; ++j) v[j] = p[j];
  return v;
}

// B-matrix 32x16 f16 tile loaded from a pre-transposed (NxK row-major) weight:
// lanes 0-15 col N=lane hold K k0..k0+15; lanes 16-31 hold K k0+16..k0+31.
__device__ __forceinline__ v16h load_b_f16(const _Float16* base, int ld,
                                           int n0, int k0, int lane) {
  const int hf = lane >> 4, n = lane & 15;
  const _Float16* p = base + (size_t)(n0 + n) * ld + k0 + hf * 16;
  v16h v;
#pragma unroll
  for (int j = 0; j < 16; ++j) v[j] = p[j];
  return v;
}

__device__ __forceinline__ v8f wmma16(v16h a, v16h b, v8f c) {
  return __builtin_amdgcn_wmma_f32_16x16x32_f16(
      /*neg_a=*/false, a, /*neg_b=*/false, b,
      /*c_mod=*/(short)0, c, /*reuse_a=*/false, /*reuse_b=*/false);
}

// ---- small helper kernels --------------------------------------------------

// src: K x N f32 row-major  ->  dst: N x K f16 row-major (transpose+convert)
__global__ void cvt_transpose_f16(const float* __restrict__ src,
                                  _Float16* __restrict__ dst,
                                  int K, int N) {
  long idx = (long)blockIdx.x * blockDim.x + threadIdx.x;
  long total = (long)K * N;
  if (idx >= total) return;
  int n = (int)(idx % N);
  int k = (int)(idx / N);
  dst[(long)n * K + k] = (_Float16)src[idx];
}

__global__ void pos_sigmoid_kernel(const float* __restrict__ pos_emb,
                                   const float* __restrict__ pos_w,
                                   const float* __restrict__ pos_b,
                                   float* __restrict__ posw) {
  int l = blockIdx.x * blockDim.x + threadIdx.x;
  if (l >= Ll) return;
  float s = pos_b[0];
#pragma unroll
  for (int p = 0; p < 16; ++p) s += pos_emb[l * 16 + p] * pos_w[p];
  posw[l] = 1.0f / (1.0f + __expf(-s));
}

// ---- Phase A: gather -> MLP -> LN -> ks/ke projections ---------------------
// 128 threads = 4 waves; each wave owns a 16-token tile; 64 KB LDS per wave
// (e:16x512 f16, u:16x1024 f16, x/h:16x512 f16) => 256 KB dynamic LDS per WG.

__global__ void stage_a_kernel(const int* __restrict__ seq,
                               const float* __restrict__ embed,
                               const _Float16* __restrict__ w1t,  // 1024x512
                               const float* __restrict__ b1,
                               const _Float16* __restrict__ w2t,  // 512x1024
                               const float* __restrict__ b2,
                               const float* __restrict__ ln_g,
                               const float* __restrict__ ln_b,
                               const _Float16* __restrict__ semt, // 256x512
                               const float* __restrict__ sem_b,
                               const _Float16* __restrict__ epit, // 256x512
                               const float* __restrict__ epi_b,
                               float* __restrict__ ks_all,        // (B*L,256)
                               float* __restrict__ ke_all) {
  extern __shared__ char smem[];
  const int tid = threadIdx.x, wid = tid >> 5, lane = tid & 31;
  _Float16* wbase = (_Float16*)smem + (size_t)wid * 32768;
  _Float16* e_lds  = wbase;                 // 8192 halfs  (16x512)
  _Float16* u_lds  = wbase + 8192;          // 16384 halfs (16x1024)
  _Float16* xh_lds = wbase + 8192 + 16384;  // 8192 halfs  (16x512)
  const int tok0 = blockIdx.x * 64 + wid * 16;

  // gather embedding rows, cvt f32->f16 into LDS
  {
    const int r = lane & 15, hf = lane >> 4;
    const int id = seq[tok0 + r];
    const float* src = embed + (size_t)id * Hh + hf * 256;
    _Float16* dst = e_lds + r * Hh + hf * 256;
    for (int j = 0; j < 256; ++j) dst[j] = (_Float16)src[j];
  }
  __syncthreads();

  // GEMM1: u = relu(e @ w1 + b1)   [16x512 * 512x1024]
  for (int nt = 0; nt < 64; ++nt) {
    v8f acc = {};
    for (int kt = 0; kt < 16; ++kt) {
      v16h a = load_a_f16(e_lds, Hh, 0, kt * 32, lane);
      v16h b = load_b_f16(w1t, Hh, nt * 16, kt * 32, lane);
      acc = wmma16(a, b, acc);
    }
    const int col = nt * 16 + (lane & 15);
    const float bias = b1[col];
    const int rb = (lane >> 4) * 8;
#pragma unroll
    for (int i = 0; i < 8; ++i) {
      float u = acc[i] + bias;
      u_lds[(rb + i) * 1024 + col] = (_Float16)(u > 0.f ? u : 0.f);
    }
  }
  __syncthreads();

  // GEMM2 + residual: x = u @ w2 + b2 + e   [16x1024 * 1024x512]
  for (int nt = 0; nt < 32; ++nt) {
    v8f acc = {};
    for (int kt = 0; kt < 32; ++kt) {
      v16h a = load_a_f16(u_lds, 1024, 0, kt * 32, lane);
      v16h b = load_b_f16(w2t, 1024, nt * 16, kt * 32, lane);
      acc = wmma16(a, b, acc);
    }
    const int col = nt * 16 + (lane & 15);
    const float bias = b2[col];
    const int rb = (lane >> 4) * 8;
#pragma unroll
    for (int i = 0; i < 8; ++i) {
      float x = acc[i] + bias + (float)e_lds[(rb + i) * Hh + col];
      xh_lds[(rb + i) * Hh + col] = (_Float16)x;
    }
  }
  __syncthreads();

  // LayerNorm over H=512, in place in xh_lds
  {
    const int r = lane & 15, hf = lane >> 4;
    float s = 0.f, s2 = 0.f;
    for (int j = 0; j < 256; ++j) {
      float x = (float)xh_lds[r * Hh + hf * 256 + j];
      s += x; s2 += x * x;
    }
    s  += __shfl_xor(s, 16, 32);
    s2 += __shfl_xor(s2, 16, 32);
    const float mu  = s * (1.f / Hh);
    const float var = s2 * (1.f / Hh) - mu * mu;
    const float rstd = rsqrtf(var + 1e-5f);
    for (int j = 0; j < 256; ++j) {
      const int col = hf * 256 + j;
      float x = (float)xh_lds[r * Hh + col];
      xh_lds[r * Hh + col] = (_Float16)((x - mu) * rstd * ln_g[col] + ln_b[col]);
    }
  }
  __syncthreads();

  // GEMM3/4: ks = h @ sem_w + sem_b ; ke = h @ epi_w + epi_b  [16x512 * 512x256]
  for (int sel = 0; sel < 2; ++sel) {
    const _Float16* wt = sel ? epit : semt;
    const float* bb = sel ? epi_b : sem_b;
    float* outp = sel ? ke_all : ks_all;
    for (int nt = 0; nt < 16; ++nt) {
      v8f acc = {};
      for (int kt = 0; kt < 16; ++kt) {
        v16h a = load_a_f16(xh_lds, Hh, 0, kt * 32, lane);
        v16h b = load_b_f16(wt, Hh, nt * 16, kt * 32, lane);
        acc = wmma16(a, b, acc);
      }
      const int col = nt * 16 + (lane & 15);
      const float bias = bb[col];
      const int rb = (lane >> 4) * 8;
#pragma unroll
      for (int i = 0; i < 8; ++i)
        outp[(size_t)(tok0 + rb + i) * HALF + col] = acc[i] + bias;
    }
  }
}

// ---- Phase B: sequential delta-rule scan -----------------------------------
// 128 WGs (b,sel); M (256 x MSTRIDE f32 ~= 260 KB) lives in dynamic LDS.
// One thread per row; float4 row traffic (ds_*_b128, 2-way-floor banking via
// the 1040B padded stride); k_{t+1} global load software-pipelined behind the
// step-t matvec/update.

__global__ void scan_kernel(const float* __restrict__ ks_all,
                            const float* __restrict__ ke_all,
                            const float* __restrict__ posw,
                            _Float16* __restrict__ cat) {
  extern __shared__ float sm[];
  float* M   = sm;                        // 256 * MSTRIDE floats
  float* kv  = sm + 256 * MSTRIDE;        // 256
  float* red = sm + 256 * MSTRIDE + 256;  // 8 (+pad)
  const int tid = threadIdx.x;
  const int lane = tid & 31, wid = tid >> 5;
  const int b = blockIdx.x >> 1;
  const int sel = blockIdx.x & 1;
  const float* kall = (sel ? ke_all : ks_all) + (size_t)b * Ll * HALF;

  // zero state
  {
    float4* M4 = (float4*)M;
    const int n4 = 256 * MSTRIDE / 4;
    for (int idx = tid; idx < n4; idx += 256)
      M4[idx] = make_float4(0.f, 0.f, 0.f, 0.f);
  }
  __syncthreads();

  const int i = tid;  // this thread owns row i of M
  float4* Mrow = (float4*)(M + i * MSTRIDE);   // 64 float4 per row
  const float4* kv4 = (const float4*)kv;

  float cur = kall[tid];  // k_0[tid]
  for (int t = 0; t < Ll - 1; ++t) {
    kv[tid] = cur;
    __syncthreads();
    // pipeline: issue k_{t+1} load now, consume after the step (t+1 <= 1023)
    float nxt = kall[(size_t)(t + 1) * HALF + tid];
    // kk = dot(k,k) from registers
    float s = cur * cur;
#pragma unroll
    for (int off = 16; off > 0; off >>= 1) s += __shfl_down(s, off, 32);
    if (lane == 0) red[wid] = s;
    __syncthreads();
    float kk = 0.f;
#pragma unroll
    for (int w = 0; w < 8; ++w) kk += red[w];
    // vp = M k  (row i), b128 traffic
    float vp = 0.f;
    for (int c = 0; c < 64; ++c) {
      float4 m = Mrow[c];
      float4 k4 = kv4[c];
      vp += m.x * k4.x + m.y * k4.y + m.z * k4.z + m.w * k4.w;
    }
    float d = cur - vp / (kk + 1e-6f);
    if (sel) d *= posw[t];
    // M += d * k^T  (row i), b128 traffic
    for (int c = 0; c < 64; ++c) {
      float4 m = Mrow[c];
      float4 k4 = kv4[c];
      m.x += d * k4.x; m.y += d * k4.y; m.z += d * k4.z; m.w += d * k4.w;
      Mrow[c] = m;
    }
    __syncthreads();
    cur = nxt;
  }

  // readout with the last key (cur == k_{L-1}): c = M k_last
  kv[tid] = cur;
  __syncthreads();
  float c = 0.f;
  for (int cc = 0; cc < 64; ++cc) {
    float4 m = Mrow[cc];
    float4 k4 = kv4[cc];
    c += m.x * k4.x + m.y * k4.y + m.z * k4.z + m.w * k4.w;
  }
  cat[b * 512 + sel * HALF + i] = (_Float16)c;
}

// ---- Phase C: out = cat @ out_w + out_b  [64x512 * 512x32000] --------------
// 2000 WGs over N; 4 waves per WG, one 16-row m-tile each; memory-bound on the
// 32.8 MB f16 transposed weight (~1.4 us at 23.3 TB/s).

__global__ void out_gemm_kernel(const _Float16* __restrict__ cat,  // 64x512
                                const _Float16* __restrict__ outt, // 32000x512
                                const float* __restrict__ out_b,
                                float* __restrict__ out) {
  const int tid = threadIdx.x, wid = tid >> 5, lane = tid & 31;
  const int n0 = blockIdx.x * 16;
  const int r0 = wid * 16;
  v8f acc = {};
  for (int kt = 0; kt < 16; ++kt) {
    if (kt + 1 < 16)  // keep the L2->WGP pipe primed (global_prefetch_b8)
      __builtin_prefetch(outt + (size_t)(n0 + (lane & 15)) * Hh + (kt + 1) * 32, 0, 3);
    v16h a = load_a_f16(cat, Hh, r0, kt * 32, lane);
    v16h b = load_b_f16(outt, Hh, n0, kt * 32, lane);
    acc = wmma16(a, b, acc);
  }
  const int col = n0 + (lane & 15);
  const float bias = out_b[col];
  const int rb = (lane >> 4) * 8;
#pragma unroll
  for (int i = 0; i < 8; ++i)
    out[(size_t)(r0 + rb + i) * Vv + col] = acc[i] + bias;
}

// ---------------------------------------------------------------------------

extern "C" void kernel_launch(void* const* d_in, const int* in_sizes, int n_in,
                              void* d_out, int out_size, void* d_ws, size_t ws_size,
                              hipStream_t stream) {
  const int*   seq     = (const int*)  d_in[0];
  const float* embed   = (const float*)d_in[1];
  const float* w1      = (const float*)d_in[2];
  const float* b1      = (const float*)d_in[3];
  const float* w2      = (const float*)d_in[4];
  const float* b2      = (const float*)d_in[5];
  const float* ln_g    = (const float*)d_in[6];
  const float* ln_b    = (const float*)d_in[7];
  const float* sem_w   = (const float*)d_in[8];
  const float* sem_b   = (const float*)d_in[9];
  const float* epi_w   = (const float*)d_in[10];
  const float* epi_b   = (const float*)d_in[11];
  const float* out_w   = (const float*)d_in[12];
  const float* out_b   = (const float*)d_in[13];
  const float* pos_emb = (const float*)d_in[14];
  const float* pos_w   = (const float*)d_in[15];
  const float* pos_b   = (const float*)d_in[16];
  (void)in_sizes; (void)n_in; (void)out_size; (void)ws_size;

  // Workspace carving (peak ~137 MB). outt aliases ks_all (dead after scan).
  char* ws = (char*)d_ws;
  size_t o = 0;
  float*    ks_all = (float*)(ws + o);    o += (size_t)Bb * Ll * HALF * 4; // 64 MB
  float*    ke_all = (float*)(ws + o);    o += (size_t)Bb * Ll * HALF * 4; // 64 MB
  _Float16* w1t    = (_Float16*)(ws + o); o += (size_t)1024 * 512 * 2;     // 1 MB
  _Float16* w2t    = (_Float16*)(ws + o); o += (size_t)512 * 1024 * 2;     // 1 MB
  _Float16* semt   = (_Float16*)(ws + o); o += (size_t)256 * 512 * 2;
  _Float16* epit   = (_Float16*)(ws + o); o += (size_t)256 * 512 * 2;
  float*    posw   = (float*)(ws + o);    o += (size_t)Ll * 4;
  _Float16* catb   = (_Float16*)(ws + o); o += (size_t)Bb * 512 * 2;
  _Float16* outt   = (_Float16*)ks_all;   // alias: 32.8 MB <= 64 MB

  // 1) weight convert+transpose (f32 KxN -> f16 NxK)
  {
    long t;
    t = 512L * 1024; cvt_transpose_f16<<<(t + 255) / 256, 256, 0, stream>>>(w1, w1t, 512, 1024);
    t = 1024L * 512; cvt_transpose_f16<<<(t + 255) / 256, 256, 0, stream>>>(w2, w2t, 1024, 512);
    t = 512L * 256;  cvt_transpose_f16<<<(t + 255) / 256, 256, 0, stream>>>(sem_w, semt, 512, 256);
    t = 512L * 256;  cvt_transpose_f16<<<(t + 255) / 256, 256, 0, stream>>>(epi_w, epit, 512, 256);
  }

  // 2) positional gate
  pos_sigmoid_kernel<<<(Ll + 255) / 256, 256, 0, stream>>>(pos_emb, pos_w, pos_b, posw);

  // 3) Phase A: 65536 tokens / (4 waves x 16 tokens) = 1024 WGs, 256 KB LDS
  stage_a_kernel<<<1024, 128, 4 * 32768 * sizeof(_Float16), stream>>>(
      seq, embed, w1t, b1, w2t, b2, ln_g, ln_b,
      semt, sem_b, epit, epi_b, ks_all, ke_all);

  // 4) Phase B: 128 WGs (b, sel), padded state matrix in ~261 KB LDS
  scan_kernel<<<2 * Bb, 256, (256 * MSTRIDE + 256 + 64) * sizeof(float), stream>>>(
      ks_all, ke_all, posw, catb);

  // 5) out_w convert+transpose AFTER the scan (aliases ks_all region)
  {
    long t = 512L * 32000;
    cvt_transpose_f16<<<(t + 255) / 256, 256, 0, stream>>>(out_w, outt, 512, 32000);
  }

  // 6) Phase C: final GEMM -> d_out (64 x 32000 f32)
  out_gemm_kernel<<<Vv / 16, 128, 0, stream>>>(catb, outt, out_b, (float*)d_out);
}
